// SioConvPSLayer_4896262717933
// MI455X (gfx1250) — matrix operand order repaired and checked by reading, not compiled, not run on hardware
//
#include <hip/hip_runtime.h>
#include <hip/hip_bf16.h>

// ---------------------------------------------------------------------------
// SioConvPSLayer for MI455X (gfx1250, wave32, WMMA + async-to-LDS pipeline)
// B=4, L=4096, D=1024; M = B*L = 16384
// ---------------------------------------------------------------------------

typedef _Float16 h4   __attribute__((ext_vector_type(4)));
typedef _Float16 h8   __attribute__((ext_vector_type(8)));
typedef _Float16 v16h __attribute__((ext_vector_type(16)));
typedef float    v8f  __attribute__((ext_vector_type(8)));
typedef float    f4   __attribute__((ext_vector_type(4)));

#define MDIM 16384
#define DDIM 1024
#define LDIM 4096
#define BDIM 4
#define NCHUNK 32
#define CHUNK 128
#define NT (DDIM / 32)          // 32 k-tiles
#define WSZ (DDIM * DDIM)       // halves per weight plane (1M)

// ---- CDNA5 primitives ------------------------------------------------------

// Async DMA: 16 bytes global -> LDS per lane, tracked by ASYNCcnt.
__device__ __forceinline__ void async_copy16(unsigned int lds_addr, const void* gaddr) {
  asm volatile("global_load_async_to_lds_b128 %0, %1, off"
               :: "v"(lds_addr), "v"(gaddr) : "memory");
}
__device__ __forceinline__ void wait_async() {
  asm volatile("s_wait_asynccnt 0" ::: "memory");
}

__device__ __forceinline__ v16h ldfrag(const _Float16* p, int o0, int o1) {
  h8 x = *(const h8*)(p + o0);
  h8 y = *(const h8*)(p + o1);
  return __builtin_shufflevector(x, y, 0,1,2,3,4,5,6,7,8,9,10,11,12,13,14,15);
}

__device__ __forceinline__ v8f wmma_f16(v16h a, v16h b, v8f c) {
  return __builtin_amdgcn_wmma_f32_16x16x32_f16(false, a, false, b, (short)0, c,
                                                false, false);
}

__device__ __forceinline__ float logsig(float u) {
  return fminf(u, 0.f) - log1pf(expf(-fabsf(u)));
}

__device__ __forceinline__ float logaddexp(float x, float y) {
  float mx = fmaxf(x, y), mn = fminf(x, y);
  if (mx == -INFINITY) return -INFINITY;
  return mx + log1pf(expf(mn - mx));
}

// Issue async copies for one 128x32 B-tile (NP f16 planes) into LDS buffer.
// 256 threads: thread -> (row = tid>>1, k-half = (tid&1)*16 halves); 2 x b128 each.
template <int NP>
__device__ __forceinline__ void issueB(const _Float16* const* planes, int n0, int k0,
                                       _Float16* BbBase, int buf, int tid) {
  const int row   = tid >> 1;
  const int halfo = (tid & 1) * 16;  // halves
#pragma unroll
  for (int p = 0; p < NP; ++p) {
    const _Float16* g = planes[p] + (size_t)(n0 + row) * DDIM + k0 + halfo;
    _Float16* l = BbBase + ((size_t)(buf * NP + p)) * 4096 + row * 32 + halfo;
    unsigned int laddr = (unsigned int)(uintptr_t)l;
    async_copy16(laddr,      g);
    async_copy16(laddr + 16, g + 8);
  }
}

// Convert 8 staged f32 A-values to f16 hi/lo and store to LDS buffer.
__device__ __forceinline__ void cvtStoreA(_Float16* Ab, int buf, int arow, int akq,
                                          f4 a0, f4 a1) {
  _Float16* dH = Ab + (size_t)(buf * 2 + 0) * 2048 + arow * 32 + akq;
  _Float16* dL = Ab + (size_t)(buf * 2 + 1) * 2048 + arow * 32 + akq;
#pragma unroll
  for (int i = 0; i < 4; ++i) {
    float v0 = a0[i], v1 = a1[i];
    _Float16 h0 = (_Float16)v0, h1 = (_Float16)v1;
    dH[i]     = h0;  dH[i + 4] = h1;
    dL[i]     = (_Float16)(v0 - (float)h0);
    dL[i + 4] = (_Float16)(v1 - (float)h1);
  }
}

// ---- kernel 0: split a weight matrix into f16 hi/lo planes ----------------

__global__ __launch_bounds__(256) void wsplit_kernel(const float* __restrict__ W,
                                                     _Float16* __restrict__ hi,
                                                     _Float16* __restrict__ lo) {
  const int i = (blockIdx.x * 256 + threadIdx.x) * 4;
  f4 v = *(const f4*)(W + i);
  h4 vh, vl;
#pragma unroll
  for (int j = 0; j < 4; ++j) {
    _Float16 h = (_Float16)v[j];
    vh[j] = h;
    vl[j] = (_Float16)(v[j] - (float)h);
  }
  *(h4*)(hi + i) = vh;
  *(h4*)(lo + i) = vl;
}

// ---- kernel 1: fused 3-way projection + activation epilogue ---------------
// Block tile 64(M) x 128(N), 8 waves (2x4), wave subtile 32x32 per matrix.
// Double-buffered LDS; B tiles arrive via async-to-LDS DMA (pre-split planes),
// A tile for step t+1 is register-staged while step t's WMMAs run.

__global__ __launch_bounds__(256) void proj3_kernel(
    const float* __restrict__ X,
    const _Float16* __restrict__ wsp,   // 6 planes: zH zL dH dL gH gL
    const float* __restrict__ bz, const float* __restrict__ bd,
    const float* __restrict__ bg,
    float* __restrict__ a_out, float* __restrict__ o_out,
    float* __restrict__ gate_out) {
  extern __shared__ __attribute__((aligned(16))) _Float16 smem[];
  _Float16* Ab = smem;          // 2 bufs x 2 planes x 64*32   = 8192 halves
  _Float16* Bb = smem + 8192;   // 2 bufs x 6 planes x 128*32  = 49152 halves

  const int tid  = threadIdx.x;
  const int lane = tid & 31;
  const int w    = tid >> 5;
  const int wm   = w >> 2, wn = w & 3;
  const int m0   = blockIdx.x * 64;
  const int n0   = blockIdx.y * 128;

  const _Float16* planes[6] = {wsp,           wsp + WSZ,     wsp + 2 * WSZ,
                               wsp + 3 * WSZ, wsp + 4 * WSZ, wsp + 5 * WSZ};

  const v8f vzero = {0.f, 0.f, 0.f, 0.f, 0.f, 0.f, 0.f, 0.f};
  v8f acc[3][2][2];
#pragma unroll
  for (int i = 0; i < 3; ++i)
#pragma unroll
    for (int j = 0; j < 2; ++j)
#pragma unroll
      for (int k = 0; k < 2; ++k) acc[i][j][k] = vzero;

  const int arow = tid >> 2, akq = (tid & 3) * 8;
  const float* Arow = X + (size_t)(m0 + arow) * DDIM + akq;

  // prologue: stage tile 0
  issueB<6>(planes, n0, 0, Bb, 0, tid);
  {
    f4 a0 = *(const f4*)(Arow);
    f4 a1 = *(const f4*)(Arow + 4);
    cvtStoreA(Ab, 0, arow, akq, a0, a1);
  }
  wait_async();
  __syncthreads();

  for (int t = 0; t < NT; ++t) {
    const int cur = t & 1, nxt = cur ^ 1;
    const bool hasNext = (t + 1) < NT;
    f4 a0, a1;
    if (hasNext) {
      issueB<6>(planes, n0, (t + 1) * 32, Bb, nxt, tid);
      a0 = *(const f4*)(Arow + (t + 1) * 32);
      a1 = *(const f4*)(Arow + (t + 1) * 32 + 4);
    }

    // ---- compute tile t ----
    {
      const int r16 = lane & 15, hsel = lane >> 4;
      const int ak0 = hsel * 8;   // A lane K pattern {0-7,16-23} / {8-15,24-31}
      const int bk0 = hsel * 8;   // B lane K pattern {0-7,16-23} / {8-15,24-31}
      v16h aH[2], aL[2];
#pragma unroll
      for (int mf = 0; mf < 2; ++mf) {
        const int row = (wm * 32 + mf * 16 + r16) * 32;
        aH[mf] = ldfrag(Ab + (cur * 2 + 0) * 2048 + row, ak0, ak0 + 16);
        aL[mf] = ldfrag(Ab + (cur * 2 + 1) * 2048 + row, ak0, ak0 + 16);
      }
#pragma unroll
      for (int mat = 0; mat < 3; ++mat) {
#pragma unroll
        for (int nf = 0; nf < 2; ++nf) {
          const int row = (wn * 32 + nf * 16 + r16) * 32;
          v16h bH = ldfrag(Bb + (cur * 6 + mat * 2 + 0) * 4096 + row, bk0, bk0 + 16);
          v16h bL = ldfrag(Bb + (cur * 6 + mat * 2 + 1) * 4096 + row, bk0, bk0 + 16);
#pragma unroll
          for (int mf = 0; mf < 2; ++mf) {
            acc[mat][mf][nf] = wmma_f16(aH[mf], bH, acc[mat][mf][nf]);
            acc[mat][mf][nf] = wmma_f16(aH[mf], bL, acc[mat][mf][nf]);
            acc[mat][mf][nf] = wmma_f16(aL[mf], bH, acc[mat][mf][nf]);
          }
        }
      }
    }

    if (hasNext) {
      cvtStoreA(Ab, nxt, arow, akq, a0, a1);
      wait_async();
    }
    __syncthreads();
  }

  // ---- epilogue ----
  const int r16 = lane & 15, hsel = lane >> 4;
#pragma unroll
  for (int mf = 0; mf < 2; ++mf) {
#pragma unroll
    for (int nf = 0; nf < 2; ++nf) {
      const int n = n0 + wn * 32 + nf * 16 + r16;
      const float vbz = bz[n], vbd = bd[n], vbg = bg[n];
#pragma unroll
      for (int r = 0; r < 8; ++r) {
        const int m = m0 + wm * 32 + mf * 16 + hsel * 8 + r;
        const size_t idx = (size_t)m * DDIM + n;
        const float z  = acc[0][mf][nf][r] + vbz;
        const float dt = acc[1][mf][nf][r] + vbd;
        const float g  = acc[2][mf][nf][r] + vbg;
        a_out[idx]    = logsig(z) + logsig(dt);   // ln_z_da
        o_out[idx]    = logsig(-dt);              // ln_o_da
        gate_out[idx] = g / (1.f + expf(-g));     // silu (parked in y half)
      }
    }
  }
}

// ---- kernel 2: chunk-local scan (log space) -------------------------------

__global__ __launch_bounds__(256) void scan_pass1(
    const float* __restrict__ a, const float* __restrict__ o,
    float* __restrict__ chunkA, float* __restrict__ chunkLSE) {
  const int tid = blockIdx.x * blockDim.x + threadIdx.x;
  const int d = tid & (DDIM - 1);
  const int c = (tid >> 10) & (NCHUNK - 1);
  const int b = tid >> 15;
  const size_t base = ((size_t)b * LDIM + (size_t)c * CHUNK) * DDIM + d;

  float A = 0.f, m = -INFINITY, s = 0.f;
  for (int t = 0; t < CHUNK; ++t) {
    const size_t idx = base + (size_t)t * DDIM;
    A += o[idx];
    const float wv = a[idx] - A;
    if (wv <= m) {
      s += expf(wv - m);
    } else {
      s = s * expf(m - wv) + 1.f;
      m = wv;
    }
  }
  const int sidx = (b * NCHUNK + c) * DDIM + d;
  chunkA[sidx]   = A;
  chunkLSE[sidx] = (s > 0.f) ? (m + logf(s)) : -INFINITY;
}

// ---- kernel 3: cross-chunk combine ----------------------------------------

__global__ __launch_bounds__(256) void scan_pass2(
    const float* __restrict__ chunkA, const float* __restrict__ chunkLSE,
    float* __restrict__ Pbuf, float* __restrict__ Cbuf) {
  const int tid = blockIdx.x * blockDim.x + threadIdx.x;  // B*D threads
  const int d = tid & (DDIM - 1);
  const int b = tid >> 10;
  float P = 0.f, C = -INFINITY;
  for (int c = 0; c < NCHUNK; ++c) {
    const int sidx = (b * NCHUNK + c) * DDIM + d;
    Pbuf[sidx] = P;
    Cbuf[sidx] = C;
    C = logaddexp(C, chunkLSE[sidx] - P);
    P += chunkA[sidx];
  }
}

// ---- kernel 4: apply — recompute local scan, emit h -----------------------

__global__ __launch_bounds__(256) void scan_pass3(
    const float* __restrict__ a, const float* __restrict__ o,
    const float* __restrict__ Pbuf, const float* __restrict__ Cbuf,
    const float* __restrict__ hidden, float* __restrict__ h) {
  const int tid = blockIdx.x * blockDim.x + threadIdx.x;
  const int d = tid & (DDIM - 1);
  const int c = (tid >> 10) & (NCHUNK - 1);
  const int b = tid >> 15;
  const size_t base = ((size_t)b * LDIM + (size_t)c * CHUNK) * DDIM + d;
  const int sidx = (b * NCHUNK + c) * DDIM + d;

  const float P   = Pbuf[sidx];
  const float Cin = Cbuf[sidx];
  const float hid = hidden[b * DDIM + d];
  const float carry = expf(P) * (expf(Cin) + hid);

  float A = 0.f, m = -INFINITY, s = 0.f;
  for (int t = 0; t < CHUNK; ++t) {
    const size_t idx = base + (size_t)t * DDIM;
    A += o[idx];
    const float wv = a[idx] - A;
    if (wv <= m) {
      s += expf(wv - m);
    } else {
      s = s * expf(m - wv) + 1.f;
      m = wv;
    }
    h[idx] = expf(A) * carry + expf(A + m) * s;
  }
}

// ---- kernel 5: y = (h @ W_y^T + b_y) * gate -------------------------------

__global__ __launch_bounds__(256) void ygemm_kernel(
    const float* __restrict__ H,
    const _Float16* __restrict__ wsp,   // 2 planes: yH yL
    const float* __restrict__ by,
    float* __restrict__ Y) {
  extern __shared__ __attribute__((aligned(16))) _Float16 smem[];
  _Float16* Ab = smem;          // 2 x 2 x 2048  = 8192 halves
  _Float16* Bb = smem + 8192;   // 2 x 2 x 4096  = 16384 halves

  const int tid  = threadIdx.x;
  const int lane = tid & 31;
  const int w    = tid >> 5;
  const int wm   = w >> 2, wn = w & 3;
  const int m0   = blockIdx.x * 64;
  const int n0   = blockIdx.y * 128;

  const _Float16* planes[2] = {wsp, wsp + WSZ};

  const v8f vzero = {0.f, 0.f, 0.f, 0.f, 0.f, 0.f, 0.f, 0.f};
  v8f acc[2][2];
#pragma unroll
  for (int j = 0; j < 2; ++j)
#pragma unroll
    for (int k = 0; k < 2; ++k) acc[j][k] = vzero;

  const int arow = tid >> 2, akq = (tid & 3) * 8;
  const float* Arow = H + (size_t)(m0 + arow) * DDIM + akq;

  issueB<2>(planes, n0, 0, Bb, 0, tid);
  {
    f4 a0 = *(const f4*)(Arow);
    f4 a1 = *(const f4*)(Arow + 4);
    cvtStoreA(Ab, 0, arow, akq, a0, a1);
  }
  wait_async();
  __syncthreads();

  for (int t = 0; t < NT; ++t) {
    const int cur = t & 1, nxt = cur ^ 1;
    const bool hasNext = (t + 1) < NT;
    f4 a0, a1;
    if (hasNext) {
      issueB<2>(planes, n0, (t + 1) * 32, Bb, nxt, tid);
      a0 = *(const f4*)(Arow + (t + 1) * 32);
      a1 = *(const f4*)(Arow + (t + 1) * 32 + 4);
    }

    {
      const int r16 = lane & 15, hsel = lane >> 4;
      const int ak0 = hsel * 8;
      const int bk0 = hsel * 8;
      v16h aH[2], aL[2];
#pragma unroll
      for (int mf = 0; mf < 2; ++mf) {
        const int row = (wm * 32 + mf * 16 + r16) * 32;
        aH[mf] = ldfrag(Ab + (cur * 2 + 0) * 2048 + row, ak0, ak0 + 16);
        aL[mf] = ldfrag(Ab + (cur * 2 + 1) * 2048 + row, ak0, ak0 + 16);
      }
#pragma unroll
      for (int nf = 0; nf < 2; ++nf) {
        const int row = (wn * 32 + nf * 16 + r16) * 32;
        v16h bH = ldfrag(Bb + (cur * 2 + 0) * 4096 + row, bk0, bk0 + 16);
        v16h bL = ldfrag(Bb + (cur * 2 + 1) * 4096 + row, bk0, bk0 + 16);
#pragma unroll
        for (int mf = 0; mf < 2; ++mf) {
          acc[mf][nf] = wmma_f16(aH[mf], bH, acc[mf][nf]);
          acc[mf][nf] = wmma_f16(aH[mf], bL, acc[mf][nf]);
          acc[mf][nf] = wmma_f16(aL[mf], bH, acc[mf][nf]);
        }
      }
    }

    if (hasNext) {
      cvtStoreA(Ab, nxt, arow, akq, a0, a1);
      wait_async();
    }
    __syncthreads();
  }

  const int r16 = lane & 15, hsel = lane >> 4;
#pragma unroll
  for (int mf = 0; mf < 2; ++mf) {
#pragma unroll
    for (int nf = 0; nf < 2; ++nf) {
      const int n = n0 + wn * 32 + nf * 16 + r16;
      const float vby = by[n];
#pragma unroll
      for (int r = 0; r < 8; ++r) {
        const int m = m0 + wm * 32 + mf * 16 + hsel * 8 + r;
        const size_t idx = (size_t)m * DDIM + n;
        const float gate = Y[idx];
        Y[idx] = (acc[mf][nf][r] + vby) * gate;
      }
    }
  }
}

// ---------------------------------------------------------------------------

extern "C" void kernel_launch(void* const* d_in, const int* in_sizes, int n_in,
                              void* d_out, int out_size, void* d_ws, size_t ws_size,
                              hipStream_t stream) {
  (void)in_sizes; (void)n_in; (void)out_size; (void)ws_size;

  const float* x      = (const float*)d_in[0];
  const float* hidden = (const float*)d_in[1];
  const float* W_ln_z = (const float*)d_in[2];
  const float* b_ln_z = (const float*)d_in[3];
  const float* W_dt   = (const float*)d_in[4];
  const float* b_dt   = (const float*)d_in[5];
  const float* W_y    = (const float*)d_in[6];
  const float* b_y    = (const float*)d_in[7];
  const float* W_g    = (const float*)d_in[8];
  const float* b_g    = (const float*)d_in[9];

  const size_t BLD = (size_t)BDIM * LDIM * DDIM;   // 16,777,216
  const size_t S   = (size_t)BDIM * NCHUNK * DDIM; // 131,072
  float* y = (float*)d_out;
  float* h = (float*)d_out + BLD;

  float* a_ws     = (float*)d_ws;      // ln_z_da   (BLD)
  float* o_ws     = a_ws + BLD;        // ln_o_da   (BLD)
  float* chunkA   = o_ws + BLD;
  float* chunkLSE = chunkA + S;
  float* Pbuf     = chunkLSE + S;
  float* Cbuf     = Pbuf + S;
  _Float16* wsp   = (_Float16*)(Cbuf + S);  // 8 planes x WSZ halves (16 MB)

  // 0) split weights into f16 hi/lo planes (once per launch; 16 MB read)
  const int wgrid = (DDIM * DDIM) / (256 * 4);  // 1024 blocks
  wsplit_kernel<<<wgrid, 256, 0, stream>>>(W_ln_z, wsp + 0 * WSZ, wsp + 1 * WSZ);
  wsplit_kernel<<<wgrid, 256, 0, stream>>>(W_dt,   wsp + 2 * WSZ, wsp + 3 * WSZ);
  wsplit_kernel<<<wgrid, 256, 0, stream>>>(W_g,    wsp + 4 * WSZ, wsp + 5 * WSZ);
  wsplit_kernel<<<wgrid, 256, 0, stream>>>(W_y,    wsp + 6 * WSZ, wsp + 7 * WSZ);

  dim3 gemm_grid(MDIM / 64, DDIM / 128);  // 256 x 8

  // 1) fused projections: ln_z_da, ln_o_da -> ws; silu gate -> y half of out
  proj3_kernel<<<gemm_grid, 256, (8192 + 49152) * sizeof(_Float16), stream>>>(
      x, wsp, b_ln_z, b_dt, b_g, a_ws, o_ws, y);

  // 2-4) log-space chunked parallel scan -> h
  scan_pass1<<<(BDIM * NCHUNK * DDIM) / 256, 256, 0, stream>>>(a_ws, o_ws,
                                                               chunkA, chunkLSE);
  scan_pass2<<<(BDIM * DDIM) / 256, 256, 0, stream>>>(chunkA, chunkLSE, Pbuf, Cbuf);
  scan_pass3<<<(BDIM * NCHUNK * DDIM) / 256, 256, 0, stream>>>(a_ws, o_ws, Pbuf,
                                                               Cbuf, hidden, h);

  // 5) y = (h @ W_y^T + b_y) * gate   (gate read+overwritten in place)
  ygemm_kernel<<<gemm_grid, 256, (8192 + 16384) * sizeof(_Float16), stream>>>(
      h, wsp + 6 * WSZ, b_y, y);
}
